// Roberta_Sequence_decoder_88682484728502
// MI455X (gfx1250) — matrix operand order repaired and checked by reading, not compiled, not run on hardware
//
#include <hip/hip_runtime.h>
#include <hip/hip_bf16.h>

// ---------------- problem constants ----------------
#define TT 10
#define BB 16
#define SS 200
#define DD 768
#define HH 300
#define NCC 4
#define K5D (5*DD)      // 3840
#define N4H (4*HH)      // 1200
#define MTOT (TT*SS*BB) // 32000
#define HPAD 320        // K of recurrent GEMM padded to multiple of 32

typedef __bf16 bf16_t;
typedef bf16_t v16bf __attribute__((ext_vector_type(16)));
typedef float  v8f   __attribute__((ext_vector_type(8)));
typedef unsigned int uint4v __attribute__((ext_vector_type(4)));
typedef unsigned int uint8v __attribute__((ext_vector_type(8)));

__device__ __forceinline__ unsigned short f2bf(float f) {
    union { float f; unsigned int u; } x; x.f = f;
    unsigned int r = x.u + 0x7fffu + ((x.u >> 16) & 1u);
    return (unsigned short)(r >> 16);
}
__device__ __forceinline__ float bf2f(unsigned short s) {
    union { unsigned int u; float f; } x; x.u = ((unsigned int)s) << 16;
    return x.f;
}
__device__ __forceinline__ float sigm(float x) {
    return 1.0f / (1.0f + __expf(-x));
}

// A fragment: 16x32 bf16 (M x K), ISA 7.12.2 layout.
__device__ __forceinline__ v16bf load_a_frag(const unsigned short* A, int ldk, int lane) {
    union { v16bf v; unsigned short u[16]; } f;
    int l  = lane & 15;
    int hi = (lane >> 4) & 1;               // lanes 16-31 hold K base +8
    const unsigned short* r = A + l * ldk + hi * 8;
#pragma unroll
    for (int i = 0; i < 8; i++) { f.u[i] = r[i]; f.u[8 + i] = r[16 + i]; }
    return f.v;
}

// B fragment: 32x16 bf16 (K x N). Source is W[N][K] row-major (B^T);
// element (k, n) at W[n*ldk + k]; W points at row n0.
__device__ __forceinline__ v16bf load_b_frag(const unsigned short* W, int ldk, int lane) {
    union { v16bf v; unsigned short u[16]; } f;
    int l  = lane & 15;
    int hi = (lane >> 4) & 1;               // lanes 16-31 hold K base +8
    const unsigned short* r = W + (size_t)l * ldk + hi * 8;
#pragma unroll
    for (int i = 0; i < 8; i++) { f.u[i] = r[i]; f.u[8 + i] = r[16 + i]; }
    return f.v;
}

// ---------------------------------------------------------------------------
// TDM descriptor helpers (ISA 08_async_tensor §8). 2D tile load to LDS.
// group0: [1:0]=count=1, [63:32]=lds_addr, [120:64]=global byte addr,
//         [127:126]=type=2 ("image")
// group1: [17:16]=data_size(1->2B), [79:48]=tensor_dim0, [111:80]=tensor_dim1,
//         [127:112]=tile_dim0, [143:128]=tile_dim1, [207:160]=dim0_stride
// ---------------------------------------------------------------------------
__device__ __forceinline__ uint4v tdm_g0(unsigned lds_off, const void* gptr) {
    unsigned long long ga = (unsigned long long)(uintptr_t)gptr;
    uint4v g;
    g[0] = 1u;                                   // count=1, user mode
    g[1] = lds_off;                              // LDS byte address
    g[2] = (unsigned)ga;                         // global addr [31:0]
    g[3] = (unsigned)((ga >> 32) & 0x01FFFFFFu)  // global addr [56:32]
         | (2u << 30);                           // type=2
    return g;
}
__device__ __forceinline__ uint8v tdm_g1(unsigned td0, unsigned td1,
                                         unsigned tile0, unsigned tile1,
                                         unsigned long long stride0) {
    unsigned long long q0 = (1ull << 16)                       // data_size=2B
                          | ((unsigned long long)(td0 & 0xFFFFu) << 48);
    unsigned long long q1 = ((unsigned long long)(td0 >> 16) & 0xFFFFull)
                          | ((unsigned long long)td1 << 16)
                          | ((unsigned long long)tile0 << 48);
    unsigned long long q2 = (unsigned long long)tile1          // tile_dim1
                          | ((stride0 & 0xFFFFFFFFull) << 32);
    unsigned long long q3 = (stride0 >> 32) & 0xFFFFull;       // stride[47:32]
    uint8v g;
    g[0] = (unsigned)q0; g[1] = (unsigned)(q0 >> 32);
    g[2] = (unsigned)q1; g[3] = (unsigned)(q1 >> 32);
    g[4] = (unsigned)q2; g[5] = (unsigned)(q2 >> 32);
    g[6] = (unsigned)q3; g[7] = (unsigned)(q3 >> 32);
    return g;
}
__device__ __forceinline__ void tdm_issue(uint4v g0, uint8v g1) {
    uint4v z4 = {0u, 0u, 0u, 0u};
    asm volatile("tensor_load_to_lds %0, %1, %2, %3"
                 :: "s"(g0), "s"(g1), "s"(z4), "s"(z4)
                 : "memory");
}

// ---------------------------------------------------------------------------
// Kernel 1: build X = [seq | s_att | e_att | prev_s | prev_e] in bf16.
// X layout: [M = t*S*B + s*B + b][5D]. One block per (t,s,b).
// ---------------------------------------------------------------------------
__global__ __launch_bounds__(256) void build_x_kernel(
    const float* __restrict__ seq, const float* __restrict__ sp,
    const float* __restrict__ ep,  const float* __restrict__ cp,
    const float* __restrict__ bsp, const float* __restrict__ bep,
    const float* __restrict__ bcp, const float* __restrict__ um,
    const float* __restrict__ nm,  const float* __restrict__ am,
    unsigned short* __restrict__ X)
{
    int m = blockIdx.x;                 // 0..MTOT-1
    int t = m / (SS * BB);
    int r = m - t * (SS * BB);
    int s = r / BB;
    int b = r - s * BB;
    int tb = t * BB + b;

    float c0 = cp[tb * NCC + 0], c1 = cp[tb * NCC + 1], c2 = cp[tb * NCC + 2];
    float a  = am[(b * TT + t) * SS + s];
    float u  = um[(b * TT + t) * SS + s];
    float n  = nm[(b * TT + t) * SS + s];
    float wm = a * c0 + u * c1 + n * c2;
    float w_s = sp[tb * SS + s] * wm;
    float w_e = ep[tb * SS + s] * wm;

    float w_ps, w_pe;
    int tbp;
    if (t > 0) {
        tbp = (t - 1) * BB + b;
        float d0 = cp[tbp * NCC + 0], d1 = cp[tbp * NCC + 1], d2 = cp[tbp * NCC + 2];
        float a1 = am[(b * TT + (t - 1)) * SS + s];
        float u1 = um[(b * TT + (t - 1)) * SS + s];
        float n1 = nm[(b * TT + (t - 1)) * SS + s];
        float wm1 = a1 * d0 + u1 * d1 + n1 * d2;
        w_ps = sp[tbp * SS + s] * wm1;
        w_pe = ep[tbp * SS + s] * wm1;
    } else {
        tbp = b;                         // seq[0]
        float d0 = bcp[b * NCC + 0], d1 = bcp[b * NCC + 1], d2 = bcp[b * NCC + 2];
        float a0 = am[(b * TT + 0) * SS + s];
        float u0 = um[(b * TT + 0) * SS + s];
        float n0 = nm[(b * TT + 0) * SS + s];
        float wm0 = a0 * d0 + u0 * d1 + n0 * d2;
        w_ps = bsp[b * SS + s] * wm0;
        w_pe = bep[b * SS + s] * wm0;
    }

    const float* seqt = seq + ((size_t)tb  * SS + s) * DD;
    const float* seqp = seq + ((size_t)tbp * SS + s) * DD;
    unsigned short* xr = X + (size_t)m * K5D;

    for (int d = threadIdx.x; d < DD; d += 256) {
        float v  = seqt[d];
        float vp = seqp[d];
        xr[d]          = f2bf(v);
        xr[DD + d]     = f2bf(w_s  * v);
        xr[2*DD + d]   = f2bf(w_e  * v);
        xr[3*DD + d]   = f2bf(w_ps * vp);
        xr[4*DD + d]   = f2bf(w_pe * vp);
    }
}

// ---------------------------------------------------------------------------
// Kernel 2: convert weights fp32 -> bf16 (Whh padded K 300 -> 320 with zeros)
// ---------------------------------------------------------------------------
#define WIH_N (N4H * K5D)     // 4,608,000
#define WHH_N (N4H * HPAD)    // 384,000
__global__ __launch_bounds__(256) void convert_w_kernel(
    const float* __restrict__ Wihf, const float* __restrict__ Wihb,
    const float* __restrict__ Whhf, const float* __restrict__ Whhb,
    unsigned short* __restrict__ Wfb, unsigned short* __restrict__ Wbb,
    unsigned short* __restrict__ Hfb, unsigned short* __restrict__ Hbb)
{
    const int total = 2 * WIH_N + 2 * WHH_N;
    for (int i = blockIdx.x * blockDim.x + threadIdx.x; i < total;
         i += gridDim.x * blockDim.x) {
        if (i < WIH_N) {
            Wfb[i] = f2bf(Wihf[i]);
        } else if (i < 2 * WIH_N) {
            Wbb[i - WIH_N] = f2bf(Wihb[i - WIH_N]);
        } else if (i < 2 * WIH_N + WHH_N) {
            int j = i - 2 * WIH_N; int nn = j / HPAD, k = j - nn * HPAD;
            Hfb[j] = (k < HH) ? f2bf(Whhf[nn * HH + k]) : (unsigned short)0;
        } else {
            int j = i - 2 * WIH_N - WHH_N; int nn = j / HPAD, k = j - nn * HPAD;
            Hbb[j] = (k < HH) ? f2bf(Whhb[nn * HH + k]) : (unsigned short)0;
        }
    }
}

// ---------------------------------------------------------------------------
// Kernel 3: batched input GEMM. Z[dir] = X(32000x3840,bf16) * W[dir]^T -> f32
// Block tile 128(M) x 80(N), K-chunk 32, 8 waves, 5 wmma/wave/chunk.
// Tiles are streamed into double-buffered LDS by the Tensor Data Mover
// (tensor_load_to_lds, TENSORcnt), issued by wave 0 only.
// ---------------------------------------------------------------------------
#define Bb_M 128
#define Bb_N 80
#define Bb_K 32
#define NKIT (K5D / Bb_K)     // 120
__global__ __launch_bounds__(256) void gemm_input_kernel(
    const unsigned short* __restrict__ X,
    const unsigned short* __restrict__ Wf, const unsigned short* __restrict__ Wb,
    float* __restrict__ Zf, float* __restrict__ Zb)
{
    const unsigned short* W = blockIdx.z ? Wb : Wf;
    float* Z = blockIdx.z ? Zb : Zf;

    __shared__ __align__(16) unsigned short As[2 * Bb_M * Bb_K];  // 16 KB
    __shared__ __align__(16) unsigned short Bs[2 * Bb_N * Bb_K];  // 10 KB

    int tid  = threadIdx.x;
    int lane = tid & 31;
    int wv   = tid >> 5;                  // 8 waves -> rows 16*wv..16*wv+15
    int l    = lane & 15;
    int hi   = (lane >> 4) & 1;
    int m0   = blockIdx.x * Bb_M;
    int n0   = blockIdx.y * Bb_N;

    // constant parts of the tile descriptors
    uint8v g1a = tdm_g1(/*td0*/K5D, /*td1*/MTOT, /*tile0*/Bb_K, /*tile1*/Bb_M,
                        /*stride0*/(unsigned long long)K5D);
    uint8v g1b = tdm_g1(/*td0*/K5D, /*td1*/N4H,  /*tile0*/Bb_K, /*tile1*/Bb_N,
                        /*stride0*/(unsigned long long)K5D);
    unsigned as_lds = (unsigned)(uintptr_t)&As[0];
    unsigned bs_lds = (unsigned)(uintptr_t)&Bs[0];

    v8f acc[5] = {};

    // prologue: DMA tile 0 into buffer 0
    if (wv == 0) {
        tdm_issue(tdm_g0(as_lds, X + (size_t)m0 * K5D), g1a);
        tdm_issue(tdm_g0(bs_lds, W + (size_t)n0 * K5D), g1b);
    }

    for (int kk = 0; kk < NKIT; kk++) {
        int cur = kk & 1;
        if (wv == 0) {
            if (kk + 1 < NKIT) {
                int nxt = (kk + 1) & 1;
                tdm_issue(tdm_g0(as_lds + nxt * (Bb_M * Bb_K * 2),
                                 X + (size_t)m0 * K5D + (kk + 1) * Bb_K), g1a);
                tdm_issue(tdm_g0(bs_lds + nxt * (Bb_N * Bb_K * 2),
                                 W + (size_t)n0 * K5D + (kk + 1) * Bb_K), g1b);
                __builtin_amdgcn_s_wait_tensorcnt(2);  // current tile landed
            } else {
                __builtin_amdgcn_s_wait_tensorcnt(0);
            }
        }
        __syncthreads();

        const unsigned short* Ab = As + cur * (Bb_M * Bb_K);
        const unsigned short* Bbp = Bs + cur * (Bb_N * Bb_K);
        v16bf af = load_a_frag(Ab + wv * 16 * Bb_K, Bb_K, lane);
        v16bf bfr[5];
#pragma unroll
        for (int j = 0; j < 5; j++)
            bfr[j] = load_b_frag(Bbp + j * 16 * Bb_K, Bb_K, lane);
#pragma unroll
        for (int j = 0; j < 5; j++)
            acc[j] = __builtin_amdgcn_wmma_f32_16x16x32_bf16(
                false, af, false, bfr[j], (short)0, acc[j], false, false);
        __syncthreads();   // computes done before next DMA overwrites this buf
    }

#pragma unroll
    for (int j = 0; j < 5; j++) {
        int nn = n0 + j * 16 + l;
#pragma unroll
        for (int i = 0; i < 8; i++) {
            int mm = m0 + wv * 16 + hi * 8 + i;
            Z[(size_t)mm * N4H + nn] = acc[j][i];
        }
    }
}

// ---------------------------------------------------------------------------
// Kernel 4: persistent bidirectional LSTM scan. grid = 2 (dir), 480 threads
// = 15 waves; wave w owns N-tiles 5w..5w+4 of the 1200-wide gate GEMM.
// h (bf16, K padded to 320) in LDS; c in registers; preacts staged via LDS.
// ---------------------------------------------------------------------------
__global__ __launch_bounds__(480) void lstm_scan_kernel(
    const float* __restrict__ Zf, const float* __restrict__ Zb,
    const unsigned short* __restrict__ Whhf, const unsigned short* __restrict__ Whhb,
    const float* __restrict__ bihf, const float* __restrict__ bhhf,
    const float* __restrict__ bihb, const float* __restrict__ bhhb,
    float* __restrict__ Y)
{
    int dir = blockIdx.x;
    const float*          Z   = dir ? Zb   : Zf;
    const unsigned short* Whh = dir ? Whhb : Whhf;
    const float*          bih = dir ? bihb : bihf;
    const float*          bhh = dir ? bhhb : bhhf;

    __shared__ __align__(16) unsigned short h_bf[BB * HPAD];   // 10 KB
    __shared__ __align__(16) unsigned short zbuf[BB * N4H];    // 38.4 KB

    int tid  = threadIdx.x;
    int lane = tid & 31;
    int wv   = tid >> 5;          // 0..14
    int l    = lane & 15;
    int hi   = (lane >> 4) & 1;

    for (int i = tid; i < BB * HPAD; i += 480) h_bf[i] = 0;
    float c_reg[10];
#pragma unroll
    for (int j = 0; j < 10; j++) c_reg[j] = 0.0f;
    __syncthreads();

    for (int t = 0; t < TT; t++) {
        for (int step = 0; step < SS; step++) {
            int sIdx = dir ? (SS - 1 - step) : step;

            // recurrent GEMM: zbuf = h @ Whh^T  (16 x 1200, K=320)
#pragma unroll
            for (int nt = 0; nt < 5; nt++) {
                int n16 = (wv * 5 + nt) * 16;
                v8f acc = {};
#pragma unroll
                for (int kk = 0; kk < HPAD / 32; kk++) {
                    v16bf af  = load_a_frag(h_bf + kk * 32, HPAD, lane);
                    v16bf bfm = load_b_frag(Whh + (size_t)n16 * HPAD + kk * 32,
                                            HPAD, lane);
                    acc = __builtin_amdgcn_wmma_f32_16x16x32_bf16(
                        false, af, false, bfm, (short)0, acc, false, false);
                }
#pragma unroll
                for (int i = 0; i < 8; i++)
                    zbuf[(hi * 8 + i) * N4H + n16 + l] = f2bf(acc[i]);
            }
            __syncthreads();

            // elementwise gates: each thread owns 10 (b,h) pairs
            const float* Zs = Z + (size_t)((t * SS + sIdx) * BB) * N4H;
#pragma unroll
            for (int j = 0; j < 10; j++) {
                int e = j * 480 + tid;
                int b = e / HH, h = e - b * HH;
                float zi = Zs[b * N4H + h]          + bf2f(zbuf[b * N4H + h])
                         + bih[h]          + bhh[h];
                float zff= Zs[b * N4H + HH + h]     + bf2f(zbuf[b * N4H + HH + h])
                         + bih[HH + h]     + bhh[HH + h];
                float zg = Zs[b * N4H + 2*HH + h]   + bf2f(zbuf[b * N4H + 2*HH + h])
                         + bih[2*HH + h]   + bhh[2*HH + h];
                float zo = Zs[b * N4H + 3*HH + h]   + bf2f(zbuf[b * N4H + 3*HH + h])
                         + bih[3*HH + h]   + bhh[3*HH + h];
                float cg = sigm(zff) * c_reg[j] + sigm(zi) * tanhf(zg);
                float hg = sigm(zo) * tanhf(cg);
                c_reg[j] = cg;
                h_bf[b * HPAD + h] = f2bf(hg);
                // Y layout [T][B][S][2H]; backward lands at row sIdx (= yb[::-1])
                Y[(((size_t)t * BB + b) * SS + sIdx) * (2 * HH) + dir * HH + h] = hg;
            }
            __syncthreads();
        }
    }
}

// ---------------------------------------------------------------------------
// Kernel 5: classifier. logits[t,b,:] = relu(Y[t,b,:]) @ Wcls^T + bcls
// ---------------------------------------------------------------------------
#define YLEN (SS * 2 * HH)    // 120000
__global__ __launch_bounds__(256) void cls_kernel(
    const float* __restrict__ Y, const float* __restrict__ Wcls,
    const float* __restrict__ bcls, float* __restrict__ out)
{
    int tb = blockIdx.x;                  // t*B + b
    const float* y = Y + (size_t)tb * YLEN;
    int tid = threadIdx.x;
    float a0 = 0, a1 = 0, a2 = 0, a3 = 0;
    for (int j = tid; j < YLEN; j += 256) {
        float v = y[j]; v = v > 0.0f ? v : 0.0f;
        a0 += v * Wcls[j];
        a1 += v * Wcls[YLEN + j];
        a2 += v * Wcls[2 * YLEN + j];
        a3 += v * Wcls[3 * YLEN + j];
    }
    __shared__ float red[4][256];
    red[0][tid] = a0; red[1][tid] = a1; red[2][tid] = a2; red[3][tid] = a3;
    __syncthreads();
    for (int s = 128; s > 0; s >>= 1) {
        if (tid < s) {
            red[0][tid] += red[0][tid + s];
            red[1][tid] += red[1][tid + s];
            red[2][tid] += red[2][tid + s];
            red[3][tid] += red[3][tid + s];
        }
        __syncthreads();
    }
    if (tid < NCC) out[tb * NCC + tid] = red[tid][0] + bcls[tid];
}

// ---------------------------------------------------------------------------
extern "C" void kernel_launch(void* const* d_in, const int* in_sizes, int n_in,
                              void* d_out, int out_size, void* d_ws, size_t ws_size,
                              hipStream_t stream) {
    const float* seq  = (const float*)d_in[1];
    const float* sp   = (const float*)d_in[2];
    const float* ep   = (const float*)d_in[3];
    const float* cp   = (const float*)d_in[4];
    const float* bsp  = (const float*)d_in[5];
    const float* bep  = (const float*)d_in[6];
    const float* bcp  = (const float*)d_in[7];
    const float* um   = (const float*)d_in[8];
    const float* nm   = (const float*)d_in[9];
    const float* am   = (const float*)d_in[10];
    const float* Wihf = (const float*)d_in[11];
    const float* Whhf = (const float*)d_in[12];
    const float* bihf = (const float*)d_in[13];
    const float* bhhf = (const float*)d_in[14];
    const float* Wihb = (const float*)d_in[15];
    const float* Whhb = (const float*)d_in[16];
    const float* bihb = (const float*)d_in[17];
    const float* bhhb = (const float*)d_in[18];
    const float* Wcls = (const float*)d_in[19];
    const float* bcls = (const float*)d_in[20];

    char* ws = (char*)d_ws;
    size_t off = 0;
    auto take = [&](size_t bytes) -> char* {
        char* p = ws + off;
        off = (off + bytes + 255) & ~(size_t)255;
        return p;
    };
    unsigned short* Xbf  = (unsigned short*)take((size_t)MTOT * K5D * 2);
    unsigned short* Wfb  = (unsigned short*)take((size_t)WIH_N * 2);
    unsigned short* Wbb  = (unsigned short*)take((size_t)WIH_N * 2);
    unsigned short* Hfb  = (unsigned short*)take((size_t)WHH_N * 2);
    unsigned short* Hbb  = (unsigned short*)take((size_t)WHH_N * 2);
    float*          Zf   = (float*)take((size_t)MTOT * N4H * 4);
    float*          Zb   = (float*)take((size_t)MTOT * N4H * 4);
    float*          Y    = (float*)take((size_t)TT * BB * YLEN * 4);

    // 1) build fused-attention input matrix X (bf16)
    build_x_kernel<<<dim3(MTOT), 256, 0, stream>>>(
        seq, sp, ep, cp, bsp, bep, bcp, um, nm, am, Xbf);

    // 2) weights -> bf16 (Whh K-padded to 320)
    convert_w_kernel<<<dim3(4096), 256, 0, stream>>>(
        Wihf, Wihb, Whhf, Whhb, Wfb, Wbb, Hfb, Hbb);

    // 3) big batched WMMA GEMM (TDM-staged, double-buffered LDS)
    gemm_input_kernel<<<dim3(MTOT / Bb_M, N4H / Bb_N, 2), 256, 0, stream>>>(
        Xbf, Wfb, Wbb, Zf, Zb);

    // 4) sequential bidirectional LSTM scan (persistent WG per direction)
    lstm_scan_kernel<<<dim3(2), 480, 0, stream>>>(
        Zf, Zb, Hfb, Hbb, bihf, bhhf, bihb, bhhb, Y);

    // 5) classifier
    cls_kernel<<<dim3(TT * BB), 256, 0, stream>>>(Y, Wcls, bcls, (float*)d_out);
}
